// ConceptEmb_43525198578144
// MI455X (gfx1250) — matrix operand order repaired and checked by reading, not compiled, not run on hardware
//
#include <hip/hip_runtime.h>
#include <hip/hip_bf16.h>

// Problem constants (from reference)
#define BB   32
#define LSS  512
#define NN   32          // concepts per set
#define DD   128         // emb dim
#define HH   64          // attention hidden dim
#define NSETS (BB * LSS) // 16384

typedef __attribute__((ext_vector_type(16))) _Float16 v16h;
typedef __attribute__((ext_vector_type(8)))  _Float16 v8h;
typedef __attribute__((ext_vector_type(8)))  float    v8f;
typedef __attribute__((ext_vector_type(4)))  float    f32x4;  // native vec, AS-cast safe
typedef __attribute__((ext_vector_type(4)))  unsigned int u32x4;
typedef __attribute__((ext_vector_type(8)))  int          i32x8;
typedef __attribute__((ext_vector_type(4)))  int          i32x4;

// ---- forced-global (addrspace 1) memory helpers ---------------------------
// Kernel-arg pointers were being lowered to flat_load_* (which ties up DScnt
// and the LDS pipe, ISA §10.1.1); force the pure VMEM global path. Native
// ext_vector types are used (HIP_vector_type's ctors can't bind across AS).
__device__ __forceinline__ float gloadf(const float* p) {
    return *(const __attribute__((address_space(1))) float*)(__UINTPTR_TYPE__)p;
}
__device__ __forceinline__ f32x4 gloadf4(const float* p) {
    return *(const __attribute__((address_space(1))) f32x4*)(__UINTPTR_TYPE__)p;
}
__device__ __forceinline__ int gloadi(const int* p) {
    return *(const __attribute__((address_space(1))) int*)(__UINTPTR_TYPE__)p;
}
__device__ __forceinline__ void gstoref4(float* p, f32x4 v) {
    *(__attribute__((address_space(1))) f32x4*)(__UINTPTR_TYPE__)p = v;
}

// ---- WMMA fragment loaders (wave32, contiguous-K order shared by A and B) -------------------
// 16-bit A (16x32): lane holds row M=lane%16; per-lane K values are one run
// of 16 consecutive: K = k0 + hi*16 + [0..15] (matches B fragments).
// This variant reads from the f32 x tile in LDS and converts on the fly.
__device__ inline v16h load_a_frag_f32(const float* base, int row, int k0, int hi) {
    const float* p = base + row * DD + k0 + hi * 16;
    f32x4 f0 = *(const f32x4*)(p + 0);
    f32x4 f1 = *(const f32x4*)(p + 4);
    f32x4 f2 = *(const f32x4*)(p + 8);
    f32x4 f3 = *(const f32x4*)(p + 12);
    v16h r;
#pragma unroll
    for (int j = 0; j < 4; ++j) {
        r[j]      = (_Float16)f0[j];
        r[4 + j]  = (_Float16)f1[j];
        r[8 + j]  = (_Float16)f2[j];
        r[12 + j] = (_Float16)f3[j];
    }
    return r;
}

// f16 A-fragment from LDS (q rows for the scores matmul)
__device__ inline v16h load_a_frag(const _Float16* base, int ld, int row, int k0, int hi) {
    const _Float16* p = base + row * ld + k0 + hi * 16;
    v8h lo = *(const v8h*)p;
    v8h hh = *(const v8h*)(p + 8);
    v16h r;
#pragma unroll
    for (int j = 0; j < 8; ++j) { r[j] = lo[j]; r[j + 8] = hh[j]; }
    return r;
}

// 16-bit B (32x16): lane holds col N=lane%16; K = k0 + hi*16 + [0..15].
__device__ inline v16h load_b_frag_lds(const _Float16* base, int ld, int nrow, int k0, int hi) {
    const _Float16* p = base + nrow * ld + k0 + hi * 16;
    v8h lo = *(const v8h*)p;
    v8h hh = *(const v8h*)(p + 8);
    v16h r;
#pragma unroll
    for (int j = 0; j < 8; ++j) { r[j] = lo[j]; r[j + 8] = hh[j]; }
    return r;
}

// B fragment straight from the f32 weight table in global (L2/L0-resident):
// q = x @ qW^T  =>  B[K][n] = qW[n][K]; lane reads 16 consecutive f32 of row n.
__device__ inline v16h load_b_frag_w(const float* wrow16) {
    f32x4 f0 = gloadf4(wrow16 + 0);
    f32x4 f1 = gloadf4(wrow16 + 4);
    f32x4 f2 = gloadf4(wrow16 + 8);
    f32x4 f3 = gloadf4(wrow16 + 12);
    v16h r;
#pragma unroll
    for (int j = 0; j < 4; ++j) {
        r[j]      = (_Float16)f0[j];
        r[4 + j]  = (_Float16)f1[j];
        r[8 + j]  = (_Float16)f2[j];
        r[12 + j] = (_Float16)f3[j];
    }
    return r;
}

__device__ inline v8f wmma_f16(v16h a, v16h b, v8f c) {
    return __builtin_amdgcn_wmma_f32_16x16x32_f16(false, a, false, b, (short)0, c, false, false);
}

// x(32x128, f32 LDS) @ W^T(HHxDD f32 global) + b -> outh(32x64, f16 LDS).
// kc outer so A-fragments load once per kc; 8 live f32 accumulators.
__device__ inline void project_qk(const float* xf32, const float* W, const float* bvec,
                                  _Float16* outh, int l15, int hi) {
    v8f acc0[4], acc1[4];
#pragma unroll
    for (int nj = 0; nj < 4; ++nj) { v8f z = {}; acc0[nj] = z; acc1[nj] = z; }
#pragma unroll
    for (int kc = 0; kc < 4; ++kc) {
        v16h a0 = load_a_frag_f32(xf32, l15,      kc * 32, hi);
        v16h a1 = load_a_frag_f32(xf32, 16 + l15, kc * 32, hi);
#pragma unroll
        for (int nj = 0; nj < 4; ++nj) {
            v16h bf = load_b_frag_w(W + (nj * 16 + l15) * DD + kc * 32 + hi * 16);
            acc0[nj] = wmma_f16(a0, bf, acc0[nj]);
            acc1[nj] = wmma_f16(a1, bf, acc1[nj]);
        }
    }
#pragma unroll
    for (int nj = 0; nj < 4; ++nj) {
        const int col = nj * 16 + l15;
        const float bias = gloadf(bvec + col);
#pragma unroll
        for (int r = 0; r < 8; ++r) {
            const int r0 = r + hi * 8;
            outh[r0 * HH + col]        = (_Float16)(acc0[nj][r] + bias);
            outh[(16 + r0) * HH + col] = (_Float16)(acc1[nj][r] + bias);
        }
    }
}

__global__ __launch_bounds__(32)
void concept_emb_attn_kernel(const int*   __restrict__ ids_g,
                             const float* __restrict__ mask_g,
                             const float* __restrict__ times_g,
                             const float* __restrict__ emb,
                             const float* __restrict__ qW, const float* __restrict__ qb,
                             const float* __restrict__ kW, const float* __restrict__ kb,
                             const float* __restrict__ theta_t,
                             const float* __restrict__ mu_t,
                             float*       __restrict__ out) {
    __shared__ __align__(16) float    xf32[NN * DD]; // 16 KB  gathered x (TDM dest)
    __shared__ __align__(16) _Float16 qh[NN * HH];   //  4 KB
    __shared__ __align__(16) _Float16 kh[NN * HH];   //  4 KB
    __shared__ __align__(16) float    sc[NN * NN];   //  4 KB  scores / weighted attn

    const int s    = blockIdx.x;        // set index in [0, 16384)
    const int lane = threadIdx.x;       // wave32
    const int l15  = lane & 15;
    const int hi   = lane >> 4;

    const int myid = gloadi(ids_g + s * NN + lane);

    // ---- stage 0: TDM gather of 32 embedding rows into LDS ----------------
    // Tensor Data Mover gather mode (ISA §8.7): 16-bit row indices, 16 rows
    // per descriptor, row width 128 f32 (512B). Two async ops cover the set;
    // tracked by TENSORcnt and overlapped with the scalar theta/mu gathers.
    // This toolchain exposes the 6-arg builtin: (g0, g1, g2, g3, g_ext, cpol);
    // the extra int32x8 group is unused for a plain 2D gather -> zeros.
    {
        const unsigned int lds_base = (unsigned int)(__UINTPTR_TYPE__)(void*)xf32;
        const unsigned long long ga = (unsigned long long)(__UINTPTR_TYPE__)emb;
        const i32x8 gext = {};                       // unused descriptor group
#pragma unroll
        for (int op = 0; op < 2; ++op) {
            u32x4 g0;
            g0[0] = 0x80000001u;                     // count=1, gather_mode=1, 16-bit idx
            g0[1] = lds_base + op * (16 * DD * 4);   // lds_addr
            g0[2] = (unsigned int)ga;                // global_addr[31:0]
            g0[3] = (unsigned int)(ga >> 32) | (2u << 30); // addr[56:32] | type=2
            i32x8 g1;
            g1[0] = 0x00020000;                      // data_size=2 (4B elements)
            g1[1] = (int)(128u << 16);               // tensor_dim0 = 128
            g1[2] = (int)0xC3500000u;                // tensor_dim1 = 50000 (lo16 in hi half)
            g1[3] = (int)(128u << 16);               // tile_dim0 = 128 (tensor_dim1 hi16 = 0)
            g1[4] = 16;                              // tile_dim1 = 16 valid indices
            g1[5] = 128;                             // tensor_dim0_stride = 128
            g1[6] = 0;
            g1[7] = 0;
            i32x4 g2, g3;
#pragma unroll
            for (int j = 0; j < 4; ++j) {
                int a0 = __builtin_amdgcn_readlane(myid, op * 16 + 2 * j);
                int a1 = __builtin_amdgcn_readlane(myid, op * 16 + 2 * j + 1);
                g2[j] = (a0 & 0xFFFF) | (a1 << 16);
                int b0 = __builtin_amdgcn_readlane(myid, op * 16 + 8 + 2 * j);
                int b1 = __builtin_amdgcn_readlane(myid, op * 16 + 8 + 2 * j + 1);
                g3[j] = (b0 & 0xFFFF) | (b1 << 16);
            }
            __builtin_amdgcn_tensor_load_to_lds(g0, g1, g2, g3, gext, 0);
        }
    }

    // ---- per-lane scalars while the TDM runs ------------------------------
    const float mk   = gloadf(mask_g + s * NN + lane);
    const float t    = gloadf(times_g + s);
    const float mval = (mk > 0.5f) ? 0.0f : -1.0e9f;       // reference's m
    const float theta = gloadf(theta_t + myid);
    const float mu    = gloadf(mu_t + myid);
    const float sx    = 1.0f / (1.0f + __expf(-(theta - mu * t)));
    const float wlane = sx * mk;                            // pooling weight

    __builtin_amdgcn_s_wait_tensorcnt(0);                   // x tile resident in LDS
    __syncthreads();

    // ---- stage 1: Q/K projections via WMMA (64 wmma) ----------------------
    project_qk(xf32, qW, qb, qh, l15, hi);
    project_qk(xf32, kW, kb, kh, l15, hi);
    __syncthreads();

    // ---- stage 2: scores = q k^T / sqrt(H) via WMMA (8 wmma) --------------
#pragma unroll
    for (int mi = 0; mi < 2; ++mi) {
#pragma unroll
        for (int nj = 0; nj < 2; ++nj) {
            v8f acc = {};
#pragma unroll
            for (int kc = 0; kc < 2; ++kc) {
                v16h a = load_a_frag(qh, HH, mi * 16 + l15, kc * 32, hi);
                v16h b = load_b_frag_lds(kh, HH, nj * 16 + l15, kc * 32, hi);
                acc = wmma_f16(a, b, acc);
            }
#pragma unroll
            for (int r = 0; r < 8; ++r)
                sc[(mi * 16 + r + hi * 8) * NN + nj * 16 + l15] = acc[r] * 0.125f;
        }
    }
    __syncthreads();

    // ---- stage 3: faithful m*m^T mask + row softmax (lane = row) ----------
    {
        float mx = -3.4e38f;
#pragma unroll 8
        for (int j = 0; j < NN; ++j) {
            const float v = sc[lane * NN + j] + mval * __shfl(mval, j);
            sc[lane * NN + j] = v;
            mx = fmaxf(mx, v);
        }
        float sum = 0.0f;
#pragma unroll 8
        for (int j = 0; j < NN; ++j) {
            const float e = __expf(sc[lane * NN + j] - mx);
            sc[lane * NN + j] = e;
            sum += e;
        }
        const float scale = wlane / sum;   // fold pooling weight into attn row
#pragma unroll 8
        for (int j = 0; j < NN; ++j)
            sc[lane * NN + j] *= scale;    // sc[i][j] = w_i * attn[i][j]
    }
    __syncthreads();

    // ---- stage 4: pooled = (w^T attn) @ x  (folded attn@x + pooling) ------
    // vvec[m] = sum_i w_i*attn[i][m]   (lane m reduces a column of sc)
    float vv = 0.0f;
#pragma unroll 8
    for (int i = 0; i < NN; ++i) vv += sc[i * NN + lane];

    // pooled[d] = sum_m vvec[m] * x[m][d]; lane owns 4 consecutive d.
    // x read back in full f32 from the TDM-gathered tile.
    float a0 = 0.f, a1 = 0.f, a2 = 0.f, a3 = 0.f;
    const int d0 = lane * 4;
#pragma unroll 8
    for (int m = 0; m < NN; ++m) {
        const float vm = __shfl(vv, m);
        const f32x4 xv = *(const f32x4*)&xf32[m * DD + d0];
        a0 += vm * xv[0];
        a1 += vm * xv[1];
        a2 += vm * xv[2];
        a3 += vm * xv[3];
    }
    f32x4 res; res[0] = a0; res[1] = a1; res[2] = a2; res[3] = a3;
    gstoref4(out + (size_t)s * DD + d0, res);
}

extern "C" void kernel_launch(void* const* d_in, const int* in_sizes, int n_in,
                              void* d_out, int out_size, void* d_ws, size_t ws_size,
                              hipStream_t stream) {
    const int*   ids   = (const int*)  d_in[0];
    const float* mask  = (const float*)d_in[1];
    const float* times = (const float*)d_in[2];
    const float* emb   = (const float*)d_in[3];
    const float* qW    = (const float*)d_in[4];
    const float* qb    = (const float*)d_in[5];
    const float* kW    = (const float*)d_in[6];
    const float* kb    = (const float*)d_in[7];
    const float* th    = (const float*)d_in[8];
    const float* mu    = (const float*)d_in[9];
    float* out = (float*)d_out;

    (void)in_sizes; (void)n_in; (void)out_size; (void)d_ws; (void)ws_size;

    concept_emb_attn_kernel<<<dim3(NSETS), dim3(32), 0, stream>>>(
        ids, mask, times, emb, qW, qb, kW, kb, th, mu, out);
}
